// QuantizableSparseStudent_67834713473597
// MI455X (gfx1250) — matrix operand order repaired and checked by reading, not compiled, not run on hardware
//
#include <hip/hip_runtime.h>
#include <hip/hip_bf16.h>

// ---- problem constants (match reference) ----
#define B_    32
#define CIN   12
#define L_    1024
#define T_    64
#define C1_   64
#define C2_   128
#define K_    7
#define NCLS  4
#define GAIN  2.0f
#define TH1   0.5f
#define TH2   0.5f
#define INV_TAU2 (1.0f/0.9f)
// TAU1 == 1.0 => v1 <- h1 exactly each step: layer-1 LIF is stateless,
// s1_t = (conv1(x_t)*GAIN >= TH1). Layer-2 LIF (tau=0.9) carries state.

// ---- tiling ----
#define LT    64          // L tile per workgroup
#define S1W   96          // s1 tile width (6 N-tiles of 16, covers halo)
#define S1OFF 16          // s1 col c <-> global pos l0 - S1OFF + c
#define XW    104         // x tile width
#define XOFF  20          // x col j <-> global pos l0 - XOFF + j (even => 4B pairs)
#define W1K   96          // conv1 padded reduction (12 ci * 8 taps)
#define W2K   512         // conv2 padded reduction (64 c1 * 8 taps)

typedef __attribute__((ext_vector_type(16))) __bf16 v16bf;
typedef __attribute__((ext_vector_type(8)))  float  v8f;

typedef __attribute__((address_space(1))) int* gint_p;
typedef __attribute__((address_space(3))) int* lint_p;

union AF { v16bf v; unsigned int  u[8];  };
union BF { v16bf v; unsigned short h[16]; };

__device__ inline unsigned short f2bf(float f) {
  __bf16 h = (__bf16)f;
  return __builtin_bit_cast(unsigned short, h);
}

#if defined(__gfx1250__) && __has_builtin(__builtin_amdgcn_global_load_async_to_lds_b32)
#define HAVE_ASYNC_LDS 1
#endif

__device__ inline void wait_async0() {
#if __has_builtin(__builtin_amdgcn_s_wait_asynccnt)
  __builtin_amdgcn_s_wait_asynccnt(0);
#else
  asm volatile("s_wait_asynccnt 0x0" ::: "memory");
#endif
}

// --- kernel 0: seed output with fc bias ---
__global__ void init_out_kernel(const float* __restrict__ fcb, float* __restrict__ out) {
  int i = threadIdx.x;                   // 128 = B_*NCLS
  if (i < B_ * NCLS) out[i] = fcb[i & (NCLS - 1)];
}

// --- kernel 1: x [B,CIN,L,T] f32 -> xT [B,T,CIN,L] bf16 ---
__global__ void xpose_kernel(const float* __restrict__ x, unsigned short* __restrict__ xT, int n) {
  int idx = blockIdx.x * 256 + threadIdx.x;
  if (idx >= n) return;
  int t    = idx & 63;
  int l    = (idx >> 6) & 1023;
  int rest = idx >> 16;                  // b*CIN + ci
  int ci   = rest % CIN;
  int b    = rest / CIN;
  xT[(((b * T_ + t) * CIN + ci) << 10) + l] = f2bf(x[idx]);
}

// --- kernel 2: fused SNN (conv1 -> LIF1 -> conv2 -> LIF2 over t) + FC reduce ---
__global__ __launch_bounds__(256, 1)
void snn_fused_kernel(const unsigned short* __restrict__ xT,
                      const float* __restrict__ w1, const float* __restrict__ b1,
                      const float* __restrict__ w2, const float* __restrict__ b2,
                      const float* __restrict__ fcw,
                      float* __restrict__ out) {
  __shared__ unsigned short sW2[C2_ * W2K];   // 128 KB, zero-padded tap 7
  __shared__ unsigned short sW1[C1_ * W1K];   // 12 KB
  __shared__ unsigned short sX [CIN * XW];    // ~2.4 KB, per-t x tile (bf16)
  __shared__ unsigned short sS1[C1_ * S1W];   // 12 KB, per-t binary spikes (bf16)
  __shared__ float          sRed[256][NCLS];  // 4 KB, final reduction

  const int tid  = threadIdx.x;
  const int wv   = tid >> 5;          // wave 0..7 (wave32)
  const int lane = tid & 31;
  const int half = lane >> 4;         // K-half selector in WMMA layouts
  const int ncol = lane & 15;         // N column within 16-wide tile
  const int b    = blockIdx.x >> 4;
  const int l0   = (blockIdx.x & 15) << 6;

  // ---- stage weights into LDS, padded K=7 -> 8 with zero tap ----
  for (int e = tid; e < C2_ * W2K; e += 256) {
    int co = e >> 9, r = e & 511, c1 = r >> 3, k = r & 7;
    sW2[e] = (k < K_) ? f2bf(w2[(co * C1_ + c1) * K_ + k]) : (unsigned short)0;
  }
  for (int e = tid; e < C1_ * W1K; e += 256) {
    int co = e / W1K, r = e - co * W1K, ci = r >> 3, k = r & 7;
    sW1[e] = (k < K_) ? f2bf(w1[(co * CIN + ci) * K_ + k]) : (unsigned short)0;
  }
  // pre-zero x tile: halo slots (global pos outside [0,L)) are never rewritten
  for (int e = tid; e < CIN * XW; e += 256) sX[e] = 0;

  // per-lane bias rows (fixed tile ownership per wave)
  const int mt1 = wv >> 1;                 // conv1 M-tile of this wave
  float b1v[8], b2v[8];
  #pragma unroll
  for (int i = 0; i < 8; ++i) {
    b1v[i] = b1[mt1 * 16 + half * 8 + i];
    b2v[i] = b2[wv  * 16 + half * 8 + i];  // conv2 M-tile == wave id
  }

  // persistent LIF2 state in registers: wave wv owns C2 rows [wv*16, wv*16+16),
  // N-tiles q = 0..3 covering the 64-wide L tile.
  float v2a[4][8], acca[4][8];
  #pragma unroll
  for (int q = 0; q < 4; ++q)
    #pragma unroll
    for (int i = 0; i < 8; ++i) { v2a[q][i] = 0.0f; acca[q][i] = 0.0f; }

  const int ntb1 = (wv & 1) * 3;           // conv1 N-tiles for this wave

  #pragma unroll 1
  for (int t = 0; t < T_; ++t) {
    // ---- stage x tile (bf16 pairs; XOFF even => pairs never straddle [0,L)) ----
    {
      const int rowbase0 = ((b * T_ + t) * CIN) << 10;
      for (int e2 = tid; e2 < CIN * (XW / 2); e2 += 256) {
        int ci = e2 / (XW / 2), jp = (e2 - ci * (XW / 2)) * 2;
        int gl = l0 - XOFF + jp;             // even
        if (gl >= 0 && gl < L_) {
#ifdef HAVE_ASYNC_LDS
          // CDNA5 async DMA: global -> LDS without touching VGPR data path
          __builtin_amdgcn_global_load_async_to_lds_b32(
              (gint_p)(xT + rowbase0 + (ci << 10) + gl),
              (lint_p)(&sX[ci * XW + jp]),
              0, 0);
#else
          *(unsigned int*)&sX[ci * XW + jp] =
              *(const unsigned int*)&xT[rowbase0 + (ci << 10) + gl];
#endif
        }
      }
#ifdef HAVE_ASYNC_LDS
      wait_async0();
#endif
    }
    __syncthreads();   // sX ready; all prior conv2 reads of sS1 done

    // ---- conv1 (implicit-GEMM WMMA): M=64, K=96, N=96 incl. halo ----
    {
      v8f cf1[3] = {};
      #pragma unroll 1
      for (int ks = 0; ks < 3; ++ks) {
        AF af;
        // A fragment 16x32 bf16 per ISA layout (32-bit pair loads, even addrs)
        const unsigned short* row = sW1 + (mt1 * 16 + ncol) * W1K + ks * 32 + half * 8;
        #pragma unroll
        for (int j = 0; j < 4; ++j) af.u[j]     = *(const unsigned int*)(row + 2 * j);
        #pragma unroll
        for (int j = 0; j < 4; ++j) af.u[4 + j] = *(const unsigned int*)(row + 16 + 2 * j);
        // reuse A across the 3 N-tiles
        #pragma unroll
        for (int e = 0; e < 3; ++e) {
          const int nt = ntb1 + e;
          BF bf;   // im2col at load time; tap 7 reads garbage but A tap-7 weight is 0
          #pragma unroll
          for (int j = 0; j < 8; ++j) {
            int r  = ks * 32 + 2 * j + 8 * half + (j >> 2) * 8;
            int ci = r >> 3, k0 = r & 7;
            int base = ci * XW + nt * 16 + ncol + k0 + 1;  // +1: XOFF=20 vs halo 19
            bf.h[2 * j]     = sX[base];
            bf.h[2 * j + 1] = sX[base + 1];
          }
          cf1[e] = __builtin_amdgcn_wmma_f32_16x16x32_bf16(false, af.v, false, bf.v,
                                                           (short)0, cf1[e], false, false);
        }
      }
      // LIF1 (stateless): threshold, write binary spikes (bf16 {0,1}) to LDS
      #pragma unroll
      for (int e = 0; e < 3; ++e) {
        const int nt = ntb1 + e;
        #pragma unroll
        for (int i = 0; i < 8; ++i) {
          int rrow = mt1 * 16 + half * 8 + i;
          int col  = nt * 16 + ncol;
          int gp   = l0 - S1OFF + col;               // zero-pad s1 outside [0,L)
          float y  = (cf1[e][i] + b1v[i]) * GAIN;
          unsigned short s = (gp >= 0 && gp < L_ && y >= TH1) ? (unsigned short)0x3F80u
                                                              : (unsigned short)0u;
          sS1[rrow * S1W + col] = s;
        }
      }
    }
    __syncthreads();   // sS1 ready for conv2

    // ---- conv2 (implicit-GEMM WMMA): M-tile = wv, K=512 (16 steps), 4 N-tiles ----
    {
      v8f cf2[4] = {};
      #pragma unroll 2
      for (int ks = 0; ks < 16; ++ks) {
        AF af;
        const unsigned short* row = sW2 + (wv * 16 + ncol) * W2K + ks * 32 + half * 8;
        #pragma unroll
        for (int j = 0; j < 4; ++j) af.u[j]     = *(const unsigned int*)(row + 2 * j);
        #pragma unroll
        for (int j = 0; j < 4; ++j) af.u[4 + j] = *(const unsigned int*)(row + 16 + 2 * j);
        // reuse A across the 4 N-tiles
        #pragma unroll
        for (int q = 0; q < 4; ++q) {
          BF bf;
          #pragma unroll
          for (int j = 0; j < 8; ++j) {
            int r  = ks * 32 + 2 * j + 8 * half + (j >> 2) * 8;
            int c1 = r >> 3, k0 = r & 7;
            int base = c1 * S1W + q * 16 + ncol + 13 + k0;  // +S1OFF - PAD = 13
            bf.h[2 * j]     = sS1[base];
            bf.h[2 * j + 1] = sS1[base + 1];
          }
          cf2[q] = __builtin_amdgcn_wmma_f32_16x16x32_bf16(false, af.v, false, bf.v,
                                                           (short)0, cf2[q], false, false);
        }
      }
      // LIF2 update, state held in registers across all t
      #pragma unroll
      for (int q = 0; q < 4; ++q)
        #pragma unroll
        for (int i = 0; i < 8; ++i) {
          float h = (cf2[q][i] + b2v[i]) * GAIN;
          float v = v2a[q][i];
          v += (h - v) * INV_TAU2;
          float sp = (v >= TH2) ? 1.0f : 0.0f;
          acca[q][i] += sp;
          v2a[q][i] = (sp > 0.0f) ? 0.0f : v;
        }
    }
    // next-iteration __syncthreads() (after sX fill) fences sS1 reuse
  }

  // ---- fold FC: partial[cls] = sum_{c2,l in tile} fcw[cls][c2]*acc / (T*L) ----
  float asum[8];
  #pragma unroll
  for (int i = 0; i < 8; ++i)
    asum[i] = acca[0][i] + acca[1][i] + acca[2][i] + acca[3][i];
  float p[NCLS] = {0.0f, 0.0f, 0.0f, 0.0f};
  #pragma unroll
  for (int i = 0; i < 8; ++i) {
    int m = wv * 16 + half * 8 + i;
    #pragma unroll
    for (int c = 0; c < NCLS; ++c) p[c] += fcw[c * C2_ + m] * asum[i];
  }
  const float scale = 1.0f / (float)(T_ * L_);
  #pragma unroll
  for (int c = 0; c < NCLS; ++c) sRed[tid][c] = p[c] * scale;
  __syncthreads();
  for (int s = 128; s > 0; s >>= 1) {
    if (tid < s)
      #pragma unroll
      for (int c = 0; c < NCLS; ++c) sRed[tid][c] += sRed[tid + s][c];
    __syncthreads();
  }
  if (tid == 0)
    #pragma unroll
    for (int c = 0; c < NCLS; ++c) atomicAdd(&out[b * NCLS + c], sRed[0][c]);
}

extern "C" void kernel_launch(void* const* d_in, const int* in_sizes, int n_in,
                              void* d_out, int out_size, void* d_ws, size_t ws_size,
                              hipStream_t stream) {
  const float* x   = (const float*)d_in[0];
  const float* w1  = (const float*)d_in[1];
  const float* b1  = (const float*)d_in[2];
  const float* w2  = (const float*)d_in[3];
  const float* b2  = (const float*)d_in[4];
  const float* fcw = (const float*)d_in[5];
  const float* fcb = (const float*)d_in[6];
  float* out = (float*)d_out;

  unsigned short* xT = (unsigned short*)d_ws;   // [B,T,CIN,L] bf16, ~50.3 MB

  init_out_kernel<<<1, 128, 0, stream>>>(fcb, out);

  const int n = B_ * CIN * L_ * T_;             // 25,165,824
  xpose_kernel<<<(n + 255) / 256, 256, 0, stream>>>(x, xT, n);

  // one WG per (batch, 64-wide L tile): 32 * 16 = 512 WGs, 8 waves each
  snn_fused_kernel<<<B_ * (L_ / LT), 256, 0, stream>>>(xT, w1, b1, w2, b2, fcw, out);
}
